// TabM_with_PLE_41738492183152
// MI455X (gfx1250) — compile-verified, matrix operand order checked by reading, not run on hardware
//
#include <hip/hip_runtime.h>

// ---------------------------------------------------------------------------
// TabM + PLE fused forward for gfx1250 (MI455X), v2.
// Key change vs v1: 64 batch rows per workgroup (512 thr = 16 waves,
// 4 m-groups x 4 n-groups) to raise arithmetic intensity vs L2 from
// ~22 FLOP/B to ~90 FLOP/B (each weight B-fragment feeds 4 WMMAs via WGP$).
// LDS: enc 128KB (bf16, aliased with fp32 X2) + X1 64KB = 192KB (<320KB/WGP).
// ---------------------------------------------------------------------------

typedef __attribute__((ext_vector_type(8)))  float        v8f;
typedef __attribute__((ext_vector_type(16))) __bf16       v16bf;
typedef __attribute__((ext_vector_type(4)))  unsigned int u32x4;

#define BATCH 1024
#define FEAT  100
#define NBINS 10
#define D0    1000      // FEAT*NBINS
#define D0P   1024      // padded K-dim for layer 1
#define KENS  32
#define H     512
#define NT    32        // H/16 column tiles
#define MT    64        // batch rows per workgroup

__device__ __forceinline__ __bf16 f2bf(float f) {
    union { float f; unsigned u; } x; x.f = f;
    unsigned r = x.u + 0x7FFFu + ((x.u >> 16) & 1u);   // round-to-nearest-even
    union { unsigned short s; __bf16 b; } y; y.s = (unsigned short)(r >> 16);
    return y.b;
}

union V16 { v16bf v; u32x4 u4[2]; };

// ---------------------------------------------------------------------------
// Pack weight matrix W (validK x N, fp32, row-major) into bf16 WMMA
// B-fragment layout: [kc][nt][lane][16 halfs], zero-padding K beyond validK.
//   lane<16 : col=nt*16+lane,      elems 0..7 -> K {0..7},  8..15 -> K {16..23}
//   lane>=16: col=nt*16+(lane-16), elems 0..7 -> K {8..15}, 8..15 -> K {24..31}
// ---------------------------------------------------------------------------
__global__ void pack_weights(const float* __restrict__ W, __bf16* __restrict__ out,
                             int ntCount, int validK, int N, int total) {
    int idx = blockIdx.x * blockDim.x + threadIdx.x;
    if (idx >= total) return;
    int e     = idx & 15;
    int lane  = (idx >> 4) & 31;
    int frag  = idx >> 9;               // kc*ntCount + nt
    int nt    = frag % ntCount;
    int kc    = frag / ntCount;
    int col   = nt * 16 + (lane & 15);
    int khalf = lane >> 4;
    int kk    = (e < 8) ? (e + 8 * khalf) : (e + 8 + 8 * khalf);
    int kg    = kc * 32 + kk;
    float v   = (kg < validK) ? W[kg * N + col] : 0.0f;
    out[idx]  = f2bf(v);
}

// ---------------------------------------------------------------------------
// Fused main kernel: grid (16, 32) = (batch tiles of 64, ensemble k), 512 thr.
// Wave w: m-group = w>>2 (16-row tile), n-group = w&3 (8 column tiles).
// ---------------------------------------------------------------------------
__global__ __launch_bounds__(512) void tabm_fused(
    const float* __restrict__ x,     const float* __restrict__ edges,
    const float* __restrict__ r1,    const float* __restrict__ s1,
    const float* __restrict__ b1,    const float* __restrict__ r2,
    const float* __restrict__ s2,    const float* __restrict__ b2,
    const float* __restrict__ Wh,    const float* __restrict__ bh,
    const __bf16* __restrict__ wp1,  const __bf16* __restrict__ wp2,
    float* __restrict__ preds) {

    union SA {
        __bf16 enc[MT * D0P];    // 128 KB: layer-1 A (PLE * r1, bf16)
        float  x2[MT * H];       // 128 KB: layer-2 output (fp32) for head
    };
    __shared__ SA     s_a;
    __shared__ __bf16 s_x1[MT * H];  // 64 KB: relu(layer1)*r2 = layer-2 A

    const int tid    = threadIdx.x;
    const int lane   = tid & 31;
    const int wave   = tid >> 5;
    const int mgrp   = wave >> 2;          // 0..3 : which 16-row tile
    const int ngrp   = wave & 3;           // 0..3 : which 8 column tiles
    const int b0     = blockIdx.x * MT;    // batch tile start
    const int k      = blockIdx.y;         // ensemble member
    const int arow   = (lane & 15) + mgrp * 16;  // A-fragment row in tile
    const int khalf  = lane >> 4;          // which K-half this lane holds
    const int kH     = k * H;

    // ---- Phase 0: PLE encode (x * r1) -> s_a.enc, zero-pad cols 1000..1023
    for (int idx = tid; idx < MT * D0P; idx += 512) {
        int r = idx >> 10, i = idx & (D0P - 1);
        float v = 0.0f;
        if (i < D0) {
            int f = i / NBINS, j = i - f * NBINS;
            float lo = edges[f * (NBINS + 1) + j];
            float hi = edges[f * (NBINS + 1) + j + 1];
            float t  = (x[(b0 + r) * FEAT + f] - lo) / (hi - lo);
            t = t < 0.0f ? 0.0f : (t > 1.0f ? 1.0f : t);
            v = t * r1[k * D0 + i];
        }
        s_a.enc[idx] = f2bf(v);
    }
    __syncthreads();

    // ---- Phase 1: GEMM1 64x1024 @ 1024x512, 1 m-tile x 8 n-tiles per wave --
    v8f acc[8] = {};
    for (int kc = 0; kc < D0P / 32; ++kc) {
        V16 a;
        const u32x4* ap = (const u32x4*)&s_a.enc[arow * D0P + kc * 32 + khalf * 8];
        a.u4[0] = ap[0];          // K base+{0..7}
        a.u4[1] = ap[2];          // K base+{16..23}
#pragma unroll
        for (int t = 0; t < 8; ++t) {
            int nt = ngrp * 8 + t;
            V16 b;
            const u32x4* bp = (const u32x4*)&wp1[(((kc << 5) + nt) * 32 + lane) << 4];
            b.u4[0] = bp[0]; b.u4[1] = bp[1];
            acc[t] = __builtin_amdgcn_wmma_f32_16x16x32_bf16(
                false, a.v, false, b.v, (short)0, acc[t], false, false);
        }
    }
    // fused epilogue: *s1 + b1, relu, *r2 -> bf16 layer-2 A in s_x1
#pragma unroll
    for (int t = 0; t < 8; ++t) {
        int col = (ngrp * 8 + t) * 16 + (lane & 15);
        float sc = s1[kH + col], bi = b1[kH + col], rr = r2[kH + col];
#pragma unroll
        for (int j = 0; j < 8; ++j) {
            float v = acc[t][j] * sc + bi;
            v = v > 0.0f ? v : 0.0f;
            s_x1[(mgrp * 16 + j + 8 * khalf) * H + col] = f2bf(v * rr);
        }
    }
    __syncthreads();

    // ---- Phase 2: GEMM2 64x512 @ 512x512 ------------------------------------
    v8f acc2[8] = {};
    for (int kc = 0; kc < H / 32; ++kc) {
        V16 a;
        const u32x4* ap = (const u32x4*)&s_x1[arow * H + kc * 32 + khalf * 8];
        a.u4[0] = ap[0];
        a.u4[1] = ap[2];
#pragma unroll
        for (int t = 0; t < 8; ++t) {
            int nt = ngrp * 8 + t;
            V16 b;
            const u32x4* bp = (const u32x4*)&wp2[(((kc << 5) + nt) * 32 + lane) << 4];
            b.u4[0] = bp[0]; b.u4[1] = bp[1];
            acc2[t] = __builtin_amdgcn_wmma_f32_16x16x32_bf16(
                false, a.v, false, b.v, (short)0, acc2[t], false, false);
        }
    }
    // fused epilogue: *s2 + b2, relu -> fp32 in s_a.x2 (enc region is dead)
#pragma unroll
    for (int t = 0; t < 8; ++t) {
        int col = (ngrp * 8 + t) * 16 + (lane & 15);
        float sc = s2[kH + col], bi = b2[kH + col];
#pragma unroll
        for (int j = 0; j < 8; ++j) {
            float v = acc2[t][j] * sc + bi;
            s_a.x2[(mgrp * 16 + j + 8 * khalf) * H + col] = v > 0.0f ? v : 0.0f;
        }
    }
    __syncthreads();

    // ---- Phase 3: head (512 -> 1), 8 lanes per batch row --------------------
    {
        int row = tid >> 3;          // 0..63
        int seg = tid & 7;           // 64-element segment
        float a = 0.0f;
        const float* wv = Wh + kH;
        const float* xv = &s_a.x2[row * H];
        for (int h = seg * 64; h < seg * 64 + 64; ++h) a += xv[h] * wv[h];
        // reduce across the 8-lane group (xor masks < 8 stay in group)
        for (int off = 4; off > 0; off >>= 1) a += __shfl_xor(a, off, 32);
        if (seg == 0) preds[k * BATCH + b0 + row] = a + bh[k];
    }
}

// ---------------------------------------------------------------------------
// Deterministic mean over the K ensemble members.
// ---------------------------------------------------------------------------
__global__ void reduce_k(const float* __restrict__ preds, float* __restrict__ out) {
    int b = blockIdx.x * blockDim.x + threadIdx.x;
    if (b >= BATCH) return;
    float s = 0.0f;
    for (int k = 0; k < KENS; ++k) s += preds[k * BATCH + b];
    out[b] = s * (1.0f / (float)KENS);
}

extern "C" void kernel_launch(void* const* d_in, const int* in_sizes, int n_in,
                              void* d_out, int out_size, void* d_ws, size_t ws_size,
                              hipStream_t stream) {
    const float* x     = (const float*)d_in[0];
    const float* edges = (const float*)d_in[1];
    const float* r1    = (const float*)d_in[2];
    const float* W1    = (const float*)d_in[3];
    const float* s1    = (const float*)d_in[4];
    const float* b1    = (const float*)d_in[5];
    const float* r2    = (const float*)d_in[6];
    const float* W2    = (const float*)d_in[7];
    const float* s2    = (const float*)d_in[8];
    const float* b2    = (const float*)d_in[9];
    const float* Wh    = (const float*)d_in[10];
    const float* bh    = (const float*)d_in[11];
    float* out = (float*)d_out;

    char* ws = (char*)d_ws;
    __bf16* wp1  = (__bf16*)ws;                                  // 1 MB  (32*32 frags)
    __bf16* wp2  = (__bf16*)(ws + (1u << 20));                   // 512 KB (16*32 frags)
    float*  prds = (float*)(ws + (1u << 20) + (1u << 19));       // 128 KB (32 x 1024)

    // Pack W1 (K padded 1000->1024) and W2 to bf16 B-fragment layout.
    pack_weights<<<(32 * 32 * 512 + 255) / 256, 256, 0, stream>>>(W1, wp1, NT, D0, H, 32 * 32 * 512);
    pack_weights<<<(16 * 32 * 512 + 255) / 256, 256, 0, stream>>>(W2, wp2, NT, H,  H, 16 * 32 * 512);

    dim3 grid(BATCH / MT, KENS);
    tabm_fused<<<grid, 512, 0, stream>>>(x, edges, r1, s1, b1, r2, s2, b2,
                                         Wh, bh, wp1, wp2, prds);
    reduce_k<<<BATCH / 256, 256, 0, stream>>>(prds, out);
}
